// GFocalHead_Tiny_61160334295461
// MI455X (gfx1250) — matrix operand order repaired and checked by reading, not compiled, not run on hardware
//
#include <hip/hip_runtime.h>
#include <hip/hip_bf16.h>

typedef __attribute__((ext_vector_type(16))) _Float16 v16h;
typedef __attribute__((ext_vector_type(8)))  float    v8f;

#define KTOT 1152   // 9 taps * 128 in-channels
#define FEAT 128

// ---------------------------------------------------------------------------
// CDNA5 async global->LDS copy (VGLOBAL GLOBAL_LOAD_ASYNC_TO_LDS_B128, GV mode)
// VDST VGPR = LDS byte offset, VADDR = 64-bit global address. Tracked by
// ASYNCcnt; completion enforced with s_wait_asynccnt before the barrier.
// ---------------------------------------------------------------------------
__device__ __forceinline__ void async_g2l_b128(unsigned ldsoff, const void* gaddr) {
    asm volatile("global_load_async_to_lds_b128 %0, %1, off"
                 :: "v"(ldsoff), "v"(gaddr) : "memory");
}
__device__ __forceinline__ void s_wait_async0() {
    asm volatile("s_wait_asynccnt 0" ::: "memory");
}

// ---------------------------------------------------------------------------
// Weight pack: OIHW f32 [O][128][3][3] -> f16 [Npad][1152], row n = out-ch,
// col k = tap*128 + c  (B^T layout: per-n contiguous K for fragment loads)
// ---------------------------------------------------------------------------
__global__ void pack_w(const float* __restrict__ src, _Float16* __restrict__ dst,
                       int O, int Npad) {
    int i = blockIdx.x * 256 + threadIdx.x;
    if (i >= Npad * KTOT) return;
    int n = i / KTOT, k = i % KTOT;
    int tap = k >> 7, c = k & 127;
    _Float16 v = (_Float16)0.f;
    if (n < O) v = (_Float16)src[(size_t)(n * FEAT + c) * 9 + tap];
    dst[i] = v;
}

// ---------------------------------------------------------------------------
// Zero padded f16 buffer (16B vector stores); n16 = elements/8
// ---------------------------------------------------------------------------
__global__ void zero_buf(uint4* __restrict__ p, int n16) {
    int i = blockIdx.x * 256 + threadIdx.x;
    if (i < n16) { uint4 z = {0, 0, 0, 0}; p[i] = z; }
}

// ---------------------------------------------------------------------------
// Activation pack: NCHW f32 -> spatially zero-padded NHWC f16
// dst layout [B][H+2][W+2][128]; borders pre-zeroed by zero_buf.
// ---------------------------------------------------------------------------
__global__ __launch_bounds__(256)
void pack_nhwc_pad(const float* __restrict__ src, _Float16* __restrict__ dst,
                   int H, int W) {
    __shared__ float t[32][33];
    const int HW = H * W, Wp = W + 2;
    int ptiles = (HW + 31) / 32;
    int b  = blockIdx.y / ptiles;
    int pt = blockIdx.y % ptiles;
    int c0 = blockIdx.x * 32;
    #pragma unroll
    for (int yy = 0; yy < 32; yy += 8) {
        int cl = threadIdx.y + yy;
        int p  = pt * 32 + threadIdx.x;
        if (p < HW)
            t[cl][threadIdx.x] = src[((size_t)b * FEAT + c0 + cl) * HW + p];
    }
    __syncthreads();
    #pragma unroll
    for (int yy = 0; yy < 32; yy += 8) {
        int pl = threadIdx.y + yy;
        int p  = pt * 32 + pl;
        if (p < HW) {
            int py = p / W + 1, px = p % W + 1;
            dst[(((size_t)b * (H + 2) + py) * Wp + px) * FEAT + c0 + threadIdx.x] =
                (_Float16)t[threadIdx.x][pl];
        }
    }
}

// ---------------------------------------------------------------------------
// Implicit-GEMM 3x3 SAME conv via v_wmma_f32_16x16x32_f16.
// Block: 256 thr (8 waves). Tile: M=128 pixels x N=NT*16 out-ch, BK=32.
// Padded input -> no boundary predication; double-buffered LDS fed by
// async global->LDS copies overlapped with the WMMA burst; 1 barrier / K-step.
// ---------------------------------------------------------------------------
template<int NT>
__global__ __launch_bounds__(256)
void conv3x3_wmma(const _Float16* __restrict__ xpad,   // [B][H+2][W+2][128] f16
                  const _Float16* __restrict__ wpk,    // [NT*16][1152] f16 (B^T)
                  const float* __restrict__ bias,
                  const float* __restrict__ scaleptr,
                  float* __restrict__ out,             // [M][Nreal] f32
                  int H, int W, int Nreal) {
    const int HW = H * W;
    const int Wp = W + 2;
    const int m0 = blockIdx.x * 128;
    const int tid = threadIdx.x, lane = tid & 31, wv = tid >> 5;

    __shared__ _Float16 Alds[2][128 * 32];
    __shared__ _Float16 Blds[2][NT * 16 * 32];

    // A staging: each thread owns 32B (16 halves) of one pixel row
    const int arow = tid >> 1;
    const int aoff = (tid & 1) * 16;
    int m = m0 + arow;
    int b = m / HW, rm = m % HW;
    int y = rm / W, x = rm % W;
    const _Float16* abase =
        xpad + (((size_t)b * (H + 2) + y) * Wp + x) * FEAT + aoff;

    // B staging: each active thread owns 32B of one out-channel row
    const bool bload = tid * 16 < NT * 16 * 32;
    const int bn = (tid * 16) >> 5, bk = (tid * 16) & 31;
    const _Float16* bbase = wpk + (size_t)bn * KTOT + bk;

    auto issue = [&](int bufsel, int ky, int kx, int c0, int k0) {
        const _Float16* ga = abase + ((ky * Wp + kx) * FEAT + c0);
        unsigned la = (unsigned)(size_t)&Alds[bufsel][arow * 32 + aoff];
        async_g2l_b128(la, ga);
        async_g2l_b128(la + 16, ga + 8);
        if (bload) {
            const _Float16* gb = bbase + k0;
            unsigned lb = (unsigned)(size_t)&Blds[bufsel][bn * 32 + bk];
            async_g2l_b128(lb, gb);
            async_g2l_b128(lb + 16, gb + 8);
        }
    };

    v8f acc[NT];
    v8f zero = {0.f, 0.f, 0.f, 0.f, 0.f, 0.f, 0.f, 0.f};
    #pragma unroll
    for (int t = 0; t < NT; t++) acc[t] = zero;

    // prologue: stage first K-slice
    issue(0, 0, 0, 0, 0);
    s_wait_async0();
    __syncthreads();

    // fragment lane coords (ISA 16-bit A 16x32 / B 32x16 layouts)
    const int mrow = wv * 16 + (lane & 15);
    const int kb   = (lane & 16) ? 8 : 0;    // A: lanes>=16 hold K 8-15/24-31
    const int koff = (lane & 16) ? 8 : 0;    // B: lanes>=16 hold K 8-15/24-31
    const int ncol = lane & 15;

    int buf = 0;
    for (int ky = 0; ky < 3; ky++)
    for (int kx = 0; kx < 3; kx++)
    for (int cc = 0; cc < 4; cc++) {
        // issue async loads for the next K-slice into the other buffer
        int nc = cc + 1, nkx = kx, nky = ky;
        if (nc == 4) { nc = 0; nkx++; if (nkx == 3) { nkx = 0; nky++; } }
        if (nky < 3)
            issue(buf ^ 1, nky, nkx, nc * 32, (nky * 3 + nkx) * 128 + nc * 32);

        // compute on current buffer
        const _Float16* Ab = Alds[buf];
        const _Float16* Bb = Blds[buf];
        union { v16h v; uint4 u[2]; } af;
        af.u[0] = *(const uint4*)(Ab + mrow * 32 + kb);
        af.u[1] = *(const uint4*)(Ab + mrow * 32 + 16 + kb);
        union { v16h v; uint4 u[2]; } bf[NT];
        #pragma unroll
        for (int t = 0; t < NT; t++) {
            const _Float16* bp = Bb + (t * 16 + ncol) * 32 + koff;
            bf[t].u[0] = *(const uint4*)bp;
            bf[t].u[1] = *(const uint4*)(bp + 16);
        }
        #pragma unroll
        for (int t = 0; t < NT; t++)
            acc[t] = __builtin_amdgcn_wmma_f32_16x16x32_f16(
                false, af.v, false, bf[t].v, (short)0, acc[t], false, false);

        s_wait_async0();
        __syncthreads();
        buf ^= 1;
    }

    // epilogue: C 16x16 f32 layout (VGPR v: M = v + (lane>=16)*8)
    const float sc = scaleptr ? scaleptr[0] : 1.0f;
    const int mbase = m0 + wv * 16 + ((lane & 16) ? 8 : 0);
    #pragma unroll
    for (int t = 0; t < NT; t++) {
        int n = t * 16 + ncol;
        if (n < Nreal) {
            float bv = bias ? bias[n] : 0.f;
            #pragma unroll
            for (int v = 0; v < 8; v++)
                out[(size_t)(mbase + v) * Nreal + n] = (acc[t][v] + bv) * sc;
        }
    }
}

// ---------------------------------------------------------------------------
// GroupNorm stats: one block per (b, group); 8 ch/group over HW pixels
// ---------------------------------------------------------------------------
__global__ __launch_bounds__(256)
void gn_stats(const float* __restrict__ xin, float* __restrict__ mu,
              float* __restrict__ rs, int HW) {
    int bg = blockIdx.x;
    int b = bg >> 4, g = bg & 15;
    const float* base = xin + (size_t)b * HW * FEAT + g * 8;
    float s = 0.f, q = 0.f;
    for (int p = threadIdx.x; p < HW; p += 256) {
        const float* row = base + (size_t)p * FEAT;
        #pragma unroll
        for (int c = 0; c < 8; c++) { float v = row[c]; s += v; q += v * v; }
    }
    __shared__ float ss[256], qq[256];
    ss[threadIdx.x] = s; qq[threadIdx.x] = q;
    __syncthreads();
    for (int st = 128; st > 0; st >>= 1) {
        if (threadIdx.x < st) {
            ss[threadIdx.x] += ss[threadIdx.x + st];
            qq[threadIdx.x] += qq[threadIdx.x + st];
        }
        __syncthreads();
    }
    if (threadIdx.x == 0) {
        float n = (float)HW * 8.f;
        float mm = ss[0] / n;
        float var = qq[0] / n - mm * mm;
        mu[bg] = mm;
        rs[bg] = rsqrtf(var + 1e-5f);
    }
}

// ---------------------------------------------------------------------------
// GroupNorm apply + ReLU -> padded f16 NHWC (feeds next WMMA conv)
// ---------------------------------------------------------------------------
__global__ void gn_apply_pad(const float* __restrict__ xin,
                             const float* __restrict__ mu, const float* __restrict__ rs,
                             const float* __restrict__ gamma, const float* __restrict__ beta,
                             _Float16* __restrict__ dst, int H, int W, int total) {
    int i = blockIdx.x * 256 + threadIdx.x;
    if (i >= total) return;
    int c = i & 127;
    int mm = i >> 7;
    int HW = H * W;
    int b = mm / HW, p = mm % HW;
    int bg = b * 16 + (c >> 3);
    float v = (xin[i] - mu[bg]) * rs[bg] * gamma[c] + beta[c];
    int py = p / W + 1, px = p % W + 1;
    dst[(((size_t)b * (H + 2) + py) * (W + 2) + px) * FEAT + c] =
        (_Float16)(v > 0.f ? v : 0.f);
}

// ---------------------------------------------------------------------------
// Head: per anchor — 4x softmax(17) + top4 + mean -> quality MLP(20->64->1),
// sigmoid(cls)*quality, DFL expectation -> bbox with priors.
// ---------------------------------------------------------------------------
__global__ __launch_bounds__(256)
void head_kernel(const float* __restrict__ regp,   // [B*HW][68]
                 const float* __restrict__ clsp,   // [B*HW][81]
                 const float* __restrict__ c1w, const float* __restrict__ c1b,
                 const float* __restrict__ c2w, const float* __restrict__ c2b,
                 float* __restrict__ out_cls,      // [B][Atot][80]
                 float* __restrict__ out_box,      // [B][Atot][4]
                 int Bn, int H, int W, int Aoff, int Atot, float stride) {
    __shared__ float w1[64 * 20], b1[64], w2[64];
    for (int i = threadIdx.x; i < 64 * 20; i += 256) w1[i] = c1w[i];
    if (threadIdx.x < 64) { b1[threadIdx.x] = c1b[threadIdx.x]; w2[threadIdx.x] = c2w[threadIdx.x]; }
    __syncthreads();

    const int HW = H * W;
    int idx = blockIdx.x * 256 + threadIdx.x;
    if (idx >= Bn * HW) return;
    int b = idx / HW, p = idx % HW;

    const float* rp = regp + (size_t)idx * 68;
    float stat[20], ebins[4];
    #pragma unroll
    for (int f = 0; f < 4; f++) {
        float pr[17];
        float mx = -1e30f;
        #pragma unroll
        for (int r = 0; r < 17; r++) { pr[r] = rp[f * 17 + r]; mx = fmaxf(mx, pr[r]); }
        float sum = 0.f;
        #pragma unroll
        for (int r = 0; r < 17; r++) { pr[r] = __expf(pr[r] - mx); sum += pr[r]; }
        float inv = 1.f / sum, e = 0.f;
        #pragma unroll
        for (int r = 0; r < 17; r++) { pr[r] *= inv; e += pr[r] * (float)r; }
        ebins[f] = e;
        float prevV = 1e30f; int prevI = -1; float tsum = 0.f;
        #pragma unroll
        for (int t = 0; t < 4; t++) {
            float best = -1e30f; int bi = -1;
            #pragma unroll
            for (int r = 0; r < 17; r++) {
                bool elig = (pr[r] < prevV) || (pr[r] == prevV && r > prevI);
                if (elig && pr[r] > best) { best = pr[r]; bi = r; }
            }
            prevV = best; prevI = bi;
            stat[f * 5 + t] = best; tsum += best;
        }
        stat[f * 5 + 4] = tsum * 0.25f;
    }

    float acc2 = c2b[0];
    for (int o = 0; o < 64; o++) {
        float a = b1[o];
        #pragma unroll
        for (int c = 0; c < 20; c++) a += stat[c] * w1[o * 20 + c];
        a = a > 0.f ? a : 0.f;
        acc2 += a * w2[o];
    }
    float quality = 1.f / (1.f + __expf(-acc2));

    const float* cp = clsp + (size_t)idx * 81;
    float* oc = out_cls + ((size_t)b * Atot + Aoff + p) * 80;
    for (int c = 0; c < 80; c++)
        oc[c] = quality / (1.f + __expf(-cp[c]));

    float cx = (float)(p % W) * stride;
    float cy = (float)(p / W) * stride;
    float* ob = out_box + ((size_t)b * Atot + Aoff + p) * 4;
    ob[0] = cx - ebins[0] * stride;
    ob[1] = cy - ebins[1] * stride;
    ob[2] = cx + ebins[2] * stride;
    ob[3] = cy + ebins[3] * stride;
}

// ---------------------------------------------------------------------------
// Host orchestration
// ---------------------------------------------------------------------------
static inline size_t aup(size_t x) { return (x + 255) & ~(size_t)255; }

extern "C" void kernel_launch(void* const* d_in, const int* in_sizes, int n_in,
                              void* d_out, int out_size, void* d_ws, size_t ws_size,
                              hipStream_t stream) {
    (void)in_sizes; (void)n_in; (void)out_size; (void)ws_size;

    const float* xs[3]    = {(const float*)d_in[0], (const float*)d_in[1], (const float*)d_in[2]};
    const float* cls_w    = (const float*)d_in[3];
    const float* cls_g    = (const float*)d_in[4];
    const float* cls_b    = (const float*)d_in[5];
    const float* reg_w    = (const float*)d_in[6];
    const float* reg_g    = (const float*)d_in[7];
    const float* reg_b    = (const float*)d_in[8];
    const float* gflcls_w = (const float*)d_in[9];
    const float* gflcls_b = (const float*)d_in[10];
    const float* gflreg_w = (const float*)d_in[11];
    const float* gflreg_b = (const float*)d_in[12];
    const float* conf1_w  = (const float*)d_in[13];
    const float* conf1_b  = (const float*)d_in[14];
    const float* conf2_w  = (const float*)d_in[15];
    const float* conf2_b  = (const float*)d_in[16];
    const float* scales   = (const float*)d_in[17];

    const int Bn = 16;
    const int Hs[3] = {80, 40, 20}, Ws[3] = {80, 40, 20};
    const float strides[3] = {8.f, 16.f, 32.f};
    const int Atot = 8400;
    const int Aoffs[3] = {0, 6400, 8000};
    const size_t Mmax  = (size_t)Bn * 80 * 80;            // 102400 pixels
    const size_t PEmax = (size_t)Bn * 82 * 82 * FEAT;     // padded f16 elems

    // workspace layout
    char* ws = (char*)d_ws;
    size_t o = 0;
    size_t oWT = o; o += aup((size_t)24 * FEAT * KTOT * sizeof(_Float16));
    size_t oWC = o; o += aup((size_t)3 * 96 * KTOT * sizeof(_Float16));
    size_t oWR = o; o += aup((size_t)3 * 80 * KTOT * sizeof(_Float16));
    size_t oXP = o; o += aup(PEmax * sizeof(_Float16));
    size_t oF[4];
    for (int i = 0; i < 4; i++) { oF[i] = o; o += aup(PEmax * sizeof(_Float16)); }
    size_t oCT = o; o += aup(Mmax * FEAT * sizeof(float));
    size_t oMU = o; o += aup(256 * sizeof(float));
    size_t oRS = o; o += aup(256 * sizeof(float));
    size_t oGC = o; o += aup(Mmax * 81 * sizeof(float));
    size_t oGR = o; o += aup(Mmax * 68 * sizeof(float));

    _Float16* wT = (_Float16*)(ws + oWT);
    _Float16* wC = (_Float16*)(ws + oWC);
    _Float16* wR = (_Float16*)(ws + oWR);
    _Float16* xp = (_Float16*)(ws + oXP);
    _Float16* fb[4];
    for (int i = 0; i < 4; i++) fb[i] = (_Float16*)(ws + oF[i]);
    float* ctmp = (float*)(ws + oCT);
    float* mu   = (float*)(ws + oMU);
    float* rs   = (float*)(ws + oRS);
    float* gco  = (float*)(ws + oGC);
    float* gro  = (float*)(ws + oGR);

    float* out_cls = (float*)d_out;
    float* out_box = (float*)d_out + (size_t)Bn * Atot * 80;

    const int gw128 = (128 * KTOT + 255) / 256;
    const int gw96  = (96 * KTOT + 255) / 256;
    const int gw80  = (80 * KTOT + 255) / 256;

    // ---- pack all weights ----
    for (int l = 0; l < 3; l++) {
        for (int i = 0; i < 4; i++) {
            pack_w<<<gw128, 256, 0, stream>>>(
                cls_w + (size_t)(l * 4 + i) * FEAT * FEAT * 9,
                wT + (size_t)(l * 8 + i) * FEAT * KTOT, FEAT, FEAT);
            pack_w<<<gw128, 256, 0, stream>>>(
                reg_w + (size_t)(l * 4 + i) * FEAT * FEAT * 9,
                wT + (size_t)(l * 8 + 4 + i) * FEAT * KTOT, FEAT, FEAT);
        }
        pack_w<<<gw96, 256, 0, stream>>>(gflcls_w + (size_t)l * 81 * FEAT * 9,
                                         wC + (size_t)l * 96 * KTOT, 81, 96);
        pack_w<<<gw80, 256, 0, stream>>>(gflreg_w + (size_t)l * 68 * FEAT * 9,
                                         wR + (size_t)l * 80 * KTOT, 68, 80);
    }

    // ---- per-level pipeline ----
    for (int l = 0; l < 3; l++) {
        const int H = Hs[l], W = Ws[l], HW = H * W;
        const int M = Bn * HW;                       // multiple of 128
        const int gM   = M / 128;
        const int gEle = (M * FEAT + 255) / 256;
        const size_t PE = (size_t)Bn * (H + 2) * (W + 2) * FEAT;
        const int gZ = (int)((PE / 8 + 255) / 256);

        // zero padded activation buffers (borders must be 0 for this level)
        zero_buf<<<gZ, 256, 0, stream>>>((uint4*)xp, (int)(PE / 8));
        for (int i = 0; i < 4; i++)
            zero_buf<<<gZ, 256, 0, stream>>>((uint4*)fb[i], (int)(PE / 8));

        pack_nhwc_pad<<<dim3(4, Bn * ((HW + 31) / 32)), dim3(32, 8), 0, stream>>>(
            xs[l], xp, H, W);

        // cls tower (ping-pong fb[0]/fb[1]) and reg tower (fb[2]/fb[3])
        for (int tower = 0; tower < 2; tower++) {
            const _Float16* cur = xp;
            _Float16* pp[2] = {fb[tower * 2], fb[tower * 2 + 1]};
            const float* gsrc = tower ? reg_g : cls_g;
            const float* bsrc = tower ? reg_b : cls_b;
            for (int i = 0; i < 4; i++) {
                const _Float16* wconv = wT + (size_t)(l * 8 + tower * 4 + i) * FEAT * KTOT;
                conv3x3_wmma<8><<<gM, 256, 0, stream>>>(cur, wconv, nullptr, nullptr,
                                                        ctmp, H, W, FEAT);
                gn_stats<<<256, 256, 0, stream>>>(ctmp, mu, rs, HW);
                gn_apply_pad<<<gEle, 256, 0, stream>>>(ctmp, mu, rs,
                                                       gsrc + (size_t)(l * 4 + i) * FEAT,
                                                       bsrc + (size_t)(l * 4 + i) * FEAT,
                                                       pp[i & 1], H, W, M * FEAT);
                cur = pp[i & 1];
            }
        }

        // gfl heads: cls feat ends in fb[1], reg feat in fb[3]
        conv3x3_wmma<6><<<gM, 256, 0, stream>>>(fb[1], wC + (size_t)l * 96 * KTOT,
                                                gflcls_b + (size_t)l * 81, nullptr,
                                                gco, H, W, 81);
        conv3x3_wmma<5><<<gM, 256, 0, stream>>>(fb[3], wR + (size_t)l * 80 * KTOT,
                                                gflreg_b + (size_t)l * 68, scales + l,
                                                gro, H, W, 68);

        head_kernel<<<(M + 255) / 256, 256, 0, stream>>>(
            gro, gco,
            conf1_w + (size_t)l * 64 * 20, conf1_b + (size_t)l * 64,
            conf2_w + (size_t)l * 64, conf2_b + l,
            out_cls, out_box, Bn, H, W, Aoffs[l], Atot, strides[l]);
    }
}
